// DichotomicSolver_1005022347784
// MI455X (gfx1250) — compile-verified, matrix operand order checked by reading, not run on hardware
//
#include <hip/hip_runtime.h>
#include <hip/hip_bf16.h>

typedef __attribute__((ext_vector_type(16))) _Float16 v16h;
typedef __attribute__((ext_vector_type(8)))  float    v8f;

#define S_DIM    2048
#define ROWS_PB  16
#define WAVES_PB 16
#define TPB      (WAVES_PB * 32)      // 512 threads = 16 wave32
#define COLS_PW  (S_DIM / WAVES_PB)   // 128 columns per wave
#define CHUNKS   (COLS_PW / 32)       // 4 WMMA k-chunks per wave per iter
#define NITER    19                   // ceil(log2((100-0)/(2*1e-4)))
#define STEPV    1e-4f
// K = 30 -> 0.5*K = 15 used in the tanh form of sigmoid

// sigmoid(2u) = 0.5*tanh(u) + 0.5 ; prefer gfx1250 v_tanh_f32 when available
__device__ __forceinline__ float sig2(float u) {
#if __has_builtin(__builtin_amdgcn_tanhf)
    return __builtin_fmaf(0.5f, __builtin_amdgcn_tanhf(u), 0.5f);
#elif __has_builtin(__builtin_amdgcn_tanh_f32)
    return __builtin_fmaf(0.5f, __builtin_amdgcn_tanh_f32(u), 0.5f);
#else
    // sigmoid(2u) = 1/(1+exp(-2u)) = 1/(1+exp2(-2u*log2(e)))
    float e = __builtin_amdgcn_exp2f(-2.8853900817779268f * u);
    return __builtin_amdgcn_rcpf(1.0f + e);
#endif
}

__global__ __launch_bounds__(TPB)
void dichot_solver_kernel(const float* __restrict__ x, float* __restrict__ out) {
    __shared__ float s_part[WAVES_PB * ROWS_PB];  // per-wave 16-row partial sums
    __shared__ float s_m[ROWS_PB];                // broadcast m per row

    const int tid  = threadIdx.x;
    const int wv   = tid >> 5;          // wave index in block (0..15)
    const int ln   = tid & 31;          // lane in wave
    const int rrow = ln & 15;           // WMMA A-layout: lane l holds row l&15
    const int row  = (int)blockIdx.x * ROWS_PB + rrow;
    const int colB = wv * COLS_PW + (ln >> 4) * 16;  // 16 consecutive cols per lane/chunk

    // ---- preload this lane's 64 x-values into registers (read x exactly once) ----
    const float* rp = x + (size_t)row * S_DIM + colB;
    float xs[CHUNKS * 16];
#pragma unroll
    for (int c = 0; c < CHUNKS; ++c) {
#pragma unroll
        for (int q = 0; q < 4; ++q) {
            float4 v = *(const float4*)(rp + c * 32 + q * 4);
            xs[c * 16 + q * 4 + 0] = v.x;
            xs[c * 16 + q * 4 + 1] = v.y;
            xs[c * 16 + q * 4 + 2] = v.z;
            xs[c * 16 + q * 4 + 3] = v.w;
        }
    }

    // B = all-ones (f16): D[m][n] = sum_k A[m][k] + C[m][n] -> WMMA as row-reducer
    v16h bones;
#pragma unroll
    for (int i = 0; i < 16; ++i) bones[i] = (_Float16)1.0f;

    // per-row solver state, owned by threads 0..15 (row = tid)
    float lbr = 0.0f, ubr = 100.0f, mr = 0.0f;
    bool  actr = true;

    for (int it = 0; it < NITER; ++it) {
        // m_new = active ? (lb+ub)/2 : m   (threads 0..15), broadcast via LDS
        if (tid < ROWS_PB) {
            float mid = 0.5f * (lbr + ubr);
            mr = actr ? mid : mr;
            s_m[tid] = mr;
        }
        __syncthreads();

        const float mrow = s_m[rrow];
        const float c1   = 15.0f * mrow;   // 0.5*K*m

        // ---- pointwise sigmoid + WMMA ones-reduction over this wave's 128 cols ----
        v8f acc = {};
#pragma unroll
        for (int c = 0; c < CHUNKS; ++c) {
            v16h a;
#pragma unroll
            for (int i = 0; i < 16; ++i) {
                float u = __builtin_fmaf(-15.0f, xs[c * 16 + i], c1); // 0.5*K*(m-x)
                a[i] = (_Float16)sig2(u);                              // sigmoid(K*(m-x))
            }
            acc = __builtin_amdgcn_wmma_f32_16x16x32_f16(
                      false, a, false, bones, (short)0, acc, false, false);
        }

        // D columns are identical; lane 0 holds rows 0..7 in acc[0..7], lane 16 rows 8..15
        if ((ln & 15) == 0) {
            int base = wv * ROWS_PB + (ln >> 4) * 8;
#pragma unroll
            for (int j = 0; j < 8; ++j) s_part[base + j] = acc[j];
        }
        __syncthreads();

        // ---- cross-wave reduction + reference state update (threads 0..15) ----
        if (tid < ROWS_PB) {
            float sum = 0.0f;
#pragma unroll
            for (int w = 0; w < WAVES_PB; ++w) sum += s_part[w * ROWS_PB + tid];
            float Dm = __builtin_fmaf(sum, 1.0f / (float)S_DIM, -0.5f);
            float H  = sig2(15.0f * Dm);           // sigmoid(K*Dm)
            float lbn = mr  - H * (mr  - lbr);
            float ubn = ubr - H * (ubr - mr);
            if (actr) { lbr = lbn; ubr = ubn; }
            bool found = __builtin_fabsf(Dm) < STEPV;
            actr = actr && !found && ((ubr - lbr) > 2.0f * STEPV);
        }
        // next iteration's top barrier orders the s_m / s_part reuse
    }

    if (tid < ROWS_PB) out[(size_t)blockIdx.x * ROWS_PB + tid] = mr;
}

extern "C" void kernel_launch(void* const* d_in, const int* in_sizes, int n_in,
                              void* d_out, int out_size, void* d_ws, size_t ws_size,
                              hipStream_t stream) {
    (void)n_in; (void)d_ws; (void)ws_size; (void)out_size;
    const float* x = (const float*)d_in[0];
    float* out = (float*)d_out;
    const int bs = in_sizes[0] / S_DIM;          // 4096
    dim3 grid(bs / ROWS_PB), block(TPB);         // 256 blocks x 512 threads
    dichot_solver_kernel<<<grid, block, 0, stream>>>(x, out);
}
